// RadarDecoderBlock_55018531062258
// MI455X (gfx1250) — matrix operand order repaired and checked by reading, not compiled, not run on hardware
//
#include <hip/hip_runtime.h>
#include <hip/hip_bf16.h>
#include <math.h>

typedef _Float16 v16h __attribute__((ext_vector_type(16)));
typedef _Float16 v8h  __attribute__((ext_vector_type(8)));
typedef float    v8f  __attribute__((ext_vector_type(8)));

#define HH   96
#define WW   96
#define HWP  (96*96)
#define BB   8
#define TT   6
#define NCH  64
#define LLF  5
#define NEGS 0.2f

#if defined(__has_builtin)
#if __has_builtin(__builtin_amdgcn_global_load_async_to_lds_b128)
#define ASYNC_LDS 1
#endif
#if __has_builtin(__builtin_amdgcn_s_wait_asynccnt)
#define HAVE_WAIT_ASYNC 1
#endif
#endif

#ifdef ASYNC_LDS
typedef int v4i __attribute__((vector_size(16)));
typedef __attribute__((address_space(1))) v4i as1_v4i;
typedef __attribute__((address_space(3))) v4i as3_v4i;
#endif

// ---------------------------------------------------------------------------
// Weight packing kernels (run each call; f32 -> f16, GEMM-ready layouts).
// ---------------------------------------------------------------------------
__global__ void cvt_f32_f16_kernel(const float* __restrict__ s,
                                   _Float16* __restrict__ d, int n)
{
    int i = blockIdx.x * blockDim.x + threadIdx.x;
    if (i < n) d[i] = (_Float16)s[i];
}

// pad rows beyond rows_src with zeros (flow weights: 10 -> 32 rows)
__global__ void cvt_pad_rows_kernel(const float* __restrict__ s,
                                    _Float16* __restrict__ d,
                                    int rows_src, int rows_dst, int k)
{
    int i = blockIdx.x * blockDim.x + threadIdx.x;
    if (i < rows_dst * k) {
        int r = i / k;
        d[i] = (r < rows_src) ? (_Float16)s[i] : (_Float16)0.f;
    }
}

// deconv weights (576,384,4,4) -> 4 parity classes of A[m=384][k=2304] f16
__global__ void pack_deconv_kernel(const float* __restrict__ w,
                                   _Float16* __restrict__ d)
{
    int i = blockIdx.x * blockDim.x + threadIdx.x;
    if (i >= 4 * 384 * 2304) return;
    int cls = i / (384 * 2304);
    int rem = i - cls * (384 * 2304);
    int m = rem / 2304;
    int k = rem - m * 2304;
    int py = cls >> 1, px = cls & 1;
    int kyb = (py + 1) & 1, kxb = (px + 1) & 1;
    int ci = k >> 2, j = k & 3;
    int ky = kyb + ((j >> 1) << 1);
    int kx = kxb + ((j & 1) << 1);
    d[i] = (_Float16)w[((size_t)(ci * 384 + m) * 4 + ky) * 4 + kx];
}

// ---------------------------------------------------------------------------
// Shared LDS-stage helpers: As[r][40] (M x K tile), BsT[n][40] (N x K tile,
// transposed so fragments read contiguous b128). 80B rows keep 16B alignment;
// 20*r mod 64 distinct for r mod 16 -> conflict-free fragment reads.
// ---------------------------------------------------------------------------
__device__ __forceinline__ void stage_A_row16(const _Float16* gp, _Float16* lp)
{
#ifdef ASYNC_LDS
    __builtin_amdgcn_global_load_async_to_lds_b128(
        (as1_v4i*)(void*)gp, (as3_v4i*)(void*)lp, 0, 0);
#else
    *(v8h*)lp = *(const v8h*)gp;
#endif
}

__device__ __forceinline__ void wait_stage()
{
#if defined(ASYNC_LDS) && defined(HAVE_WAIT_ASYNC)
    __builtin_amdgcn_s_wait_asynccnt(0);
#endif
}

__device__ __forceinline__ v16h frag16(const _Float16* p0, const _Float16* p1)
{
    v8h lo = *(const v8h*)p0;
    v8h hi = *(const v8h*)p1;
    return __builtin_shufflevector(lo, hi, 0, 1, 2, 3, 4, 5, 6, 7,
                                   8, 9, 10, 11, 12, 13, 14, 15);
}

// ---------------------------------------------------------------------------
// Transposed-conv (k=4,s=2,p=1) as 4 parity-class GEMMs, f16 WMMA, f32 accum.
// Packed weights wAd: [cls][384][2304] f16. Workgroup 64(M) x 64(N), 4 waves
// each 32x32 (4 WMMA / K-step). N/class = 8*48*48 = 18432.
// ---------------------------------------------------------------------------
__global__ __launch_bounds__(128)
void deconv_wmma_kernel(const float* __restrict__ x,
                        const _Float16* __restrict__ wAd,
                        const float* __restrict__ bias,
                        _Float16* __restrict__ u)
{
    __shared__ _Float16 As[64][40];
    __shared__ _Float16 BsT[64][40];

    const int cls = blockIdx.z;
    const int py = cls >> 1, px = cls & 1;
    const int tileM = blockIdx.x * 64;
    const int tileN = blockIdx.y * 64;
    const _Float16* wgt = wAd + (size_t)cls * 384 * 2304;

    const int tid  = threadIdx.x;
    const int lane = tid & 31;
    const int wave = tid >> 5;
    const int wm = (wave >> 1) * 32;
    const int wn = (wave & 1) * 32;
    const int lm = lane & 15, khalf = lane >> 4;

    const int kyb = (py + 1) & 1;
    const int kxb = (px + 1) & 1;

    // hoisted per-thread B-gather state (fixed column)
    const int col = tid & 63;
    const int r0  = (tid >> 6) * 16;
    int n  = tileN + col;
    int bI = n / (48 * 48);
    int p  = n - bI * (48 * 48);
    int oyp = p / 48, oxp = p - oyp * 48;
    int iy0 = (2 * oyp + py + 1 - kyb) >> 1;
    int iy1 = iy0 - 1;
    int ix0 = (2 * oxp + px + 1 - kxb) >> 1;
    int ix1 = ix0 - 1;
    bool vy0 = (iy0 >= 0) && (iy0 < 48), vy1 = (iy1 >= 0) && (iy1 < 48);
    bool vx0 = (ix0 >= 0) && (ix0 < 48), vx1 = (ix1 >= 0) && (ix1 < 48);
    int oy0 = iy0 * 48, oy1 = iy1 * 48;
    const float* xb = x + (size_t)bI * 576 * (48 * 48);

    v8f acc[2][2];
    #pragma unroll
    for (int a = 0; a < 2; ++a)
        #pragma unroll
        for (int b = 0; b < 2; ++b)
            #pragma unroll
            for (int i = 0; i < 8; ++i) acc[a][b][i] = 0.f;

    const int K = 576 * 4;
    for (int k0 = 0; k0 < K; k0 += 32) {
        // stage A: 64 rows x 32 k; 2 x b128 per thread (async if available)
        #pragma unroll
        for (int e = 0; e < 2; ++e) {
            int id = tid + e * 128;
            int r = id >> 2, c8 = (id & 3) * 8;
            stage_A_row16(wgt + (size_t)(tileM + r) * K + k0 + c8, &As[r][c8]);
        }
        // stage B transposed: this thread's column, 16 k-rows, 2 x b128
        {
            v8h lo, hi;
            #pragma unroll
            for (int i = 0; i < 16; ++i) {
                int k = k0 + r0 + i;
                int ci = k >> 2, j = k & 3;
                bool vy = (j & 2) ? vy1 : vy0;
                bool vx = (j & 1) ? vx1 : vx0;
                int  oy = (j & 2) ? oy1 : oy0;
                int  ox = (j & 1) ? ix1 : ix0;
                float v = (vy && vx) ? xb[(size_t)ci * (48 * 48) + oy + ox] : 0.f;
                if (i < 8) lo[i] = (_Float16)v; else hi[i - 8] = (_Float16)v;
            }
            *(v8h*)&BsT[col][r0]     = lo;
            *(v8h*)&BsT[col][r0 + 8] = hi;
        }
        wait_stage();
        __syncthreads();
        v16h fa[2], fb[2];
        #pragma unroll
        for (int am = 0; am < 2; ++am)
            fa[am] = frag16(&As[wm + am * 16 + lm][khalf * 16],
                            &As[wm + am * 16 + lm][khalf * 16 + 8]);
        #pragma unroll
        for (int bn = 0; bn < 2; ++bn)
            fb[bn] = frag16(&BsT[wn + bn * 16 + lm][khalf * 16],
                            &BsT[wn + bn * 16 + lm][khalf * 16 + 8]);
        #pragma unroll
        for (int am = 0; am < 2; ++am)
            #pragma unroll
            for (int bn = 0; bn < 2; ++bn)
                acc[am][bn] = __builtin_amdgcn_wmma_f32_16x16x32_f16(
                    false, fa[am], false, fb[bn], (short)0, acc[am][bn], false, false);
        __syncthreads();
    }
    #pragma unroll
    for (int bn = 0; bn < 2; ++bn) {
        int ns = tileN + wn + bn * 16 + lm;
        int bS = ns / (48 * 48);
        int ps = ns - bS * (48 * 48);
        int oyS = ps / 48, oxS = ps - oyS * 48;
        int oy = 2 * oyS + py, ox = 2 * oxS + px;
        #pragma unroll
        for (int am = 0; am < 2; ++am)
            #pragma unroll
            for (int rr = 0; rr < 8; ++rr) {
                int m = tileM + wm + am * 16 + rr + 8 * khalf;
                float v = acc[am][bn][rr] + bias[m];
                v = (v > 0.f) ? v : NEGS * v;
                u[((size_t)(bS * 384 + m) * HH + oy) * WW + ox] = (_Float16)v;
            }
    }
}

// ---------------------------------------------------------------------------
// Generic implicit-GEMM conv (stride 1), f16 WMMA, f32 accumulate.
// Workgroup tile (2*WM) x 64; 4 waves each WM x 32.  Weights pre-packed f16,
// rows padded to a multiple of BM.  M = real row count (store mask only).
// ---------------------------------------------------------------------------
template <typename Tin, typename Tout, int WM>
__global__ __launch_bounds__(128)
void conv_wmma_kernel(const Tin* __restrict__ in, int cstr, int chan_off,
                      int KH, int KW, int pad,
                      const _Float16* __restrict__ wgt,
                      const float* __restrict__ bias,
                      int M, int K,
                      Tout* __restrict__ out, int outC,
                      int acc_flag, int act_flag)
{
    constexpr int BM = 2 * WM;
    constexpr int AM = WM / 16;
    __shared__ _Float16 As[BM][40];
    __shared__ _Float16 BsT[64][40];

    const int tileM = blockIdx.x * BM;
    const int tileN = blockIdx.y * 64;

    const int tid  = threadIdx.x;
    const int lane = tid & 31;
    const int wave = tid >> 5;
    const int wm = (wave >> 1) * WM;
    const int wn = (wave & 1) * 32;
    const int lm = lane & 15, khalf = lane >> 4;
    const int khw = KH * KW;

    const int col = tid & 63;
    const int r0  = (tid >> 6) * 16;
    int n  = tileN + col;
    int bI = n / HWP;
    int p  = n - bI * HWP;
    int yy = p / WW, xx = p - yy * WW;
    const Tin* inb = in + ((size_t)bI * cstr + chan_off) * HWP;

    v8f acc[AM][2];
    #pragma unroll
    for (int a = 0; a < AM; ++a)
        #pragma unroll
        for (int b = 0; b < 2; ++b)
            #pragma unroll
            for (int i = 0; i < 8; ++i) acc[a][b][i] = 0.f;

    for (int k0 = 0; k0 < K; k0 += 32) {
        // stage A: BM rows x 32 k; b128 chunks (async if available)
        #pragma unroll
        for (int e = 0; e < BM / 32; ++e) {
            int id = tid + e * 128;
            int r = id >> 2, c8 = (id & 3) * 8;
            stage_A_row16(wgt + (size_t)(tileM + r) * K + k0 + c8, &As[r][c8]);
        }
        if (k0 + 32 < K)
            __builtin_prefetch(&wgt[(size_t)(tileM + (tid & (BM - 1))) * K + k0 + 32], 0, 0);
        // stage B transposed: fixed column, 16 k-rows, incremental (ci,ky,kx)
        {
            int kf  = k0 + r0;
            int ci  = kf / khw;
            int rem = kf - ci * khw;
            int ky  = rem / KW;
            int kx  = rem - ky * KW;
            v8h lo, hi;
            #pragma unroll
            for (int i = 0; i < 16; ++i) {
                int iy = yy + ky - pad, ix = xx + kx - pad;
                float v = 0.f;
                if (iy >= 0 && iy < HH && ix >= 0 && ix < WW)
                    v = (float)inb[(size_t)ci * HWP + iy * WW + ix];
                if (i < 8) lo[i] = (_Float16)v; else hi[i - 8] = (_Float16)v;
                ++kx;
                if (kx == KW) {
                    kx = 0; ++ky;
                    if (ky == KH) { ky = 0; ++ci; }
                }
            }
            *(v8h*)&BsT[col][r0]     = lo;
            *(v8h*)&BsT[col][r0 + 8] = hi;
        }
        wait_stage();
        __syncthreads();
        v16h fa[AM], fb[2];
        #pragma unroll
        for (int am = 0; am < AM; ++am)
            fa[am] = frag16(&As[wm + am * 16 + lm][khalf * 16],
                            &As[wm + am * 16 + lm][khalf * 16 + 8]);
        #pragma unroll
        for (int bn = 0; bn < 2; ++bn)
            fb[bn] = frag16(&BsT[wn + bn * 16 + lm][khalf * 16],
                            &BsT[wn + bn * 16 + lm][khalf * 16 + 8]);
        #pragma unroll
        for (int am = 0; am < AM; ++am)
            #pragma unroll
            for (int bn = 0; bn < 2; ++bn)
                acc[am][bn] = __builtin_amdgcn_wmma_f32_16x16x32_f16(
                    false, fa[am], false, fb[bn], (short)0, acc[am][bn], false, false);
        __syncthreads();
    }
    #pragma unroll
    for (int bn = 0; bn < 2; ++bn) {
        int ns = tileN + wn + bn * 16 + lm;
        int bS = ns / HWP;
        int ps = ns - bS * HWP;
        #pragma unroll
        for (int am = 0; am < AM; ++am)
            #pragma unroll
            for (int rr = 0; rr < 8; ++rr) {
                int m = tileM + wm + am * 16 + rr + 8 * khalf;
                if (m < M) {
                    size_t oidx = ((size_t)bS * outC + m) * HWP + ps;
                    float v = acc[am][bn][rr] + bias[m];
                    if (acc_flag) v += (float)out[oidx];
                    if (act_flag) v = (v > 0.f) ? v : NEGS * v;
                    out[oidx] = (Tout)v;
                }
            }
    }
}

// ---------------------------------------------------------------------------
// Bilinear warp of hidden state by 5 flow fields (zeros padding,
// align_corners=False).  h read from d_out slice (t-1); h_toff<0 => h==0.
// ---------------------------------------------------------------------------
__global__ void warp_kernel(const float* __restrict__ flows,
                            const float* __restrict__ hbase,
                            int h_toff,
                            _Float16* __restrict__ warped)
{
    int e = blockIdx.x * blockDim.x + threadIdx.x;
    if (e >= BB * HWP) return;
    int bI = e / HWP;
    int p  = e - bI * HWP;
    int yy = p / WW, xx = p - yy * WW;
    const float sx = 2.f / (float)(WW - 1);
    const float sy = 2.f / (float)(HH - 1);

    for (int l = 0; l < LLF; ++l) {
        _Float16* wout = warped + ((size_t)bI * 320 + l * NCH) * HWP + p;
        if (h_toff < 0) {
            for (int c = 0; c < NCH; ++c) wout[(size_t)c * HWP] = (_Float16)0.f;
            continue;
        }
        float fx = flows[((size_t)bI * 10 + 2 * l)     * HWP + p];
        float fy = flows[((size_t)bI * 10 + 2 * l + 1) * HWP + p];
        float gx = sx * ((float)xx - fx) - 1.f;
        float gy = sy * ((float)yy - fy) - 1.f;
        float xs = ((gx + 1.f) * (float)WW - 1.f) * 0.5f;
        float ys = ((gy + 1.f) * (float)HH - 1.f) * 0.5f;
        float x0f = floorf(xs), y0f = floorf(ys);
        float wx = xs - x0f, wy = ys - y0f;
        int x0 = (int)x0f, y0 = (int)y0f;
        int x1 = x0 + 1,  y1 = y0 + 1;
        bool vx0 = (x0 >= 0) && (x0 < WW);
        bool vx1 = (x1 >= 0) && (x1 < WW);
        bool vy0 = (y0 >= 0) && (y0 < HH);
        bool vy1 = (y1 >= 0) && (y1 < HH);
        int x0c = x0 < 0 ? 0 : (x0 > WW - 1 ? WW - 1 : x0);
        int x1c = x1 < 0 ? 0 : (x1 > WW - 1 ? WW - 1 : x1);
        int y0c = y0 < 0 ? 0 : (y0 > HH - 1 ? HH - 1 : y0);
        int y1c = y1 < 0 ? 0 : (y1 > HH - 1 ? HH - 1 : y1);
        float w00 = (1.f - wx) * (1.f - wy);
        float w01 = wx * (1.f - wy);
        float w10 = (1.f - wx) * wy;
        float w11 = wx * wy;
        const float* hp = hbase + ((size_t)bI * 384 + h_toff) * HWP;
        for (int c = 0; c < NCH; ++c) {
            const float* hc = hp + (size_t)c * HWP;
            float v00 = (vx0 && vy0) ? hc[y0c * WW + x0c] : 0.f;
            float v01 = (vx1 && vy0) ? hc[y0c * WW + x1c] : 0.f;
            float v10 = (vx0 && vy1) ? hc[y1c * WW + x0c] : 0.f;
            float v11 = (vx1 && vy1) ? hc[y1c * WW + x1c] : 0.f;
            float v = v00 * w00 + v01 * w01 + v10 * w10 + v11 * w11;
            wout[(size_t)c * HWP] = (_Float16)v;
        }
    }
}

// ---------------------------------------------------------------------------
// GRU gates (i2h/h2h in f16): r,z sigmoid; m = lrelu(i2h3 + r*h2h3);
// hn = z*h + (1-z)*m.  Hidden state lives in d_out (B, T, 64, H, W).
// ---------------------------------------------------------------------------
__global__ void gates_kernel(const _Float16* __restrict__ i2h,
                             const _Float16* __restrict__ h2h,
                             float* __restrict__ dout, int t)
{
    int e = blockIdx.x * blockDim.x + threadIdx.x;
    if (e >= BB * NCH * HWP) return;
    int bI  = e / (NCH * HWP);
    int rem = e - bI * NCH * HWP;
    int c = rem / HWP;
    int p = rem - c * HWP;
    size_t base = ((size_t)bI * 192 + c) * HWP + p;
    float r = (float)i2h[base] + (float)h2h[base];
    r = 1.f / (1.f + expf(-r));
    float z = (float)i2h[base + (size_t)64 * HWP] + (float)h2h[base + (size_t)64 * HWP];
    z = 1.f / (1.f + expf(-z));
    float m = (float)i2h[base + (size_t)128 * HWP]
            + r * (float)h2h[base + (size_t)128 * HWP];
    m = (m > 0.f) ? m : NEGS * m;
    float hp = 0.f;
    if (t > 0) hp = dout[((size_t)bI * 384 + (t - 1) * NCH + c) * HWP + p];
    float hn = z * hp + (1.f - z) * m;
    dout[((size_t)bI * 384 + t * NCH + c) * HWP + p] = hn;
}

// ---------------------------------------------------------------------------
extern "C" void kernel_launch(void* const* d_in, const int* in_sizes, int n_in,
                              void* d_out, int out_size, void* d_ws, size_t ws_size,
                              hipStream_t stream)
{
    const float* x      = (const float*)d_in[0];
    const float* up_w   = (const float*)d_in[1];
    const float* up_b   = (const float*)d_in[2];
    const float* i2h_w  = (const float*)d_in[3];
    const float* i2h_b  = (const float*)d_in[4];
    const float* i2f_w  = (const float*)d_in[5];
    const float* i2f_b  = (const float*)d_in[6];
    const float* h2f_w  = (const float*)d_in[7];
    const float* h2f_b  = (const float*)d_in[8];
    const float* flow_w = (const float*)d_in[9];
    const float* flow_b = (const float*)d_in[10];
    const float* ret_w  = (const float*)d_in[11];
    const float* ret_b  = (const float*)d_in[12];
    float* out = (float*)d_out;

    char* ws = (char*)d_ws;
    _Float16* wAd    = (_Float16*)ws;  ws += sizeof(_Float16) * (size_t)4 * 384 * 2304;
    _Float16* i2f_h  = (_Float16*)ws;  ws += sizeof(_Float16) * (size_t)32 * 1600;
    _Float16* h2f_h  = (_Float16*)ws;  ws += sizeof(_Float16) * (size_t)32 * 1600;
    _Float16* flow_h = (_Float16*)ws;  ws += sizeof(_Float16) * (size_t)32 * 800;
    _Float16* i2h_h  = (_Float16*)ws;  ws += sizeof(_Float16) * (size_t)192 * 576;
    _Float16* ret_h  = (_Float16*)ws;  ws += sizeof(_Float16) * (size_t)192 * 320;
    _Float16* u      = (_Float16*)ws;  ws += sizeof(_Float16) * (size_t)BB * 384 * HWP;
    _Float16* warped = (_Float16*)ws;  ws += sizeof(_Float16) * (size_t)BB * 320 * HWP;
    _Float16* fbuf   = (_Float16*)ws;  ws += sizeof(_Float16) * (size_t)BB * 32  * HWP;
    _Float16* i2hbuf = (_Float16*)ws;  ws += sizeof(_Float16) * (size_t)BB * 192 * HWP;
    _Float16* h2hbuf = (_Float16*)ws;  ws += sizeof(_Float16) * (size_t)BB * 192 * HWP;
    float*    flows  = (float*)ws;     ws += sizeof(float)    * (size_t)BB * 10  * HWP;

    // ---- weight packing (f32 -> f16, GEMM layouts) ----
    pack_deconv_kernel<<<(4 * 384 * 2304 + 255) / 256, 256, 0, stream>>>(up_w, wAd);
    cvt_f32_f16_kernel<<<(32 * 1600 + 255) / 256, 256, 0, stream>>>(i2f_w, i2f_h, 32 * 1600);
    cvt_f32_f16_kernel<<<(32 * 1600 + 255) / 256, 256, 0, stream>>>(h2f_w, h2f_h, 32 * 1600);
    cvt_pad_rows_kernel<<<(32 * 800 + 255) / 256, 256, 0, stream>>>(flow_w, flow_h, 10, 32, 800);
    cvt_f32_f16_kernel<<<(192 * 576 + 255) / 256, 256, 0, stream>>>(i2h_w, i2h_h, 192 * 576);
    cvt_f32_f16_kernel<<<(192 * 320 + 255) / 256, 256, 0, stream>>>(ret_w, ret_h, 192 * 320);

    dim3 blk(128);
    const int nblk = (BB * HWP) / 64;   // 1152

    // upsample: deconv + leaky relu -> u (f16), viewed as (B, T*64, 96, 96)
    deconv_wmma_kernel<<<dim3(6, 288, 4), blk, 0, stream>>>(x, wAd, up_b, u);

    for (int t = 0; t < TT; ++t) {
        // f = lrelu(i2f(xt) + h2f(h))
        conv_wmma_kernel<_Float16, _Float16, 16><<<dim3(1, nblk), blk, 0, stream>>>(
            u, 384, t * NCH, 5, 5, 2, i2f_h, i2f_b, 32, 1600,
            fbuf, 32, 0, (t == 0) ? 1 : 0);
        if (t > 0) {
            conv_wmma_kernel<float, _Float16, 16><<<dim3(1, nblk), blk, 0, stream>>>(
                out, 384, (t - 1) * NCH, 5, 5, 2, h2f_h, h2f_b, 32, 1600,
                fbuf, 32, 1, 1);
        }
        // flows = conv5x5(f)  (A rows zero-padded 10 -> 32 at pack time)
        conv_wmma_kernel<_Float16, float, 16><<<dim3(1, nblk), blk, 0, stream>>>(
            fbuf, 32, 0, 5, 5, 2, flow_h, flow_b, 10, 800,
            flows, 10, 0, 0);
        // warp hidden state by 5 flows
        warp_kernel<<<(BB * HWP + 255) / 256, 256, 0, stream>>>(
            flows, out, (t > 0) ? (t - 1) * NCH : -1, warped);
        // i2h = conv3x3(xt), h2h = conv1x1(warped)
        conv_wmma_kernel<_Float16, _Float16, 32><<<dim3(3, nblk), blk, 0, stream>>>(
            u, 384, t * NCH, 3, 3, 1, i2h_h, i2h_b, 192, 576,
            i2hbuf, 192, 0, 0);
        conv_wmma_kernel<_Float16, _Float16, 32><<<dim3(3, nblk), blk, 0, stream>>>(
            warped, 320, 0, 1, 1, 0, ret_h, ret_b, 192, 320,
            h2hbuf, 192, 0, 0);
        // gates + state update, writes d_out[:, t]
        gates_kernel<<<(BB * NCH * HWP + 255) / 256, 256, 0, stream>>>(
            i2hbuf, h2hbuf, out, t);
    }
}
